// MotionNet_695784702584
// MI455X (gfx1250) — matrix operand order repaired and checked by reading, not compiled, not run on hardware
//
#include <hip/hip_runtime.h>
#include <hip/hip_bf16.h>

typedef __attribute__((ext_vector_type(16))) _Float16 v16h;
typedef __attribute__((ext_vector_type(8)))  _Float16 v8h;
typedef __attribute__((ext_vector_type(8)))  float    v8f;

// ---------------------------------------------------------------------------
// Shift-point + farthest point sampling. One workgroup per batch, blockDim=N.
// q[b] = pts[b+1]  (b%16 != 15)   else pts[b] + (pts[b]-pts[b-15])/15
// ---------------------------------------------------------------------------
template<int N, int NP>
__global__ void k_shift_fps(const float* __restrict__ pts, float* __restrict__ qsel) {
  const int b = blockIdx.x;
  const int t = threadIdx.x;
  __shared__ float rv[N];
  __shared__ int   ri[N];
  __shared__ float bc[3];
  const int l = b & 15;
  float px, py, pz;
  if (l != 15) {
    const float* p = pts + (size_t)((b + 1) * N + t) * 3;
    px = p[0]; py = p[1]; pz = p[2];
  } else {
    const float* p  = pts + (size_t)(b * N + t) * 3;
    const float* p0 = pts + (size_t)((b - 15) * N + t) * 3;
    const float s = 1.0f / 15.0f;
    px = p[0] + (p[0] - p0[0]) * s;
    py = p[1] + (p[1] - p0[1]) * s;
    pz = p[2] + (p[2] - p0[2]) * s;
  }
  float md = 1e10f;
  int last = 0;
  if (t == 0) { float* o = qsel + (size_t)b * NP * 3; o[0] = px; o[1] = py; o[2] = pz; }
  for (int j = 1; j < NP; ++j) {
    if (t == last) { bc[0] = px; bc[1] = py; bc[2] = pz; }
    __syncthreads();
    float dx = px - bc[0], dy = py - bc[1], dz = pz - bc[2];
    md = fminf(md, dx * dx + dy * dy + dz * dz);
    rv[t] = md; ri[t] = t;
    __syncthreads();
    for (int s = N >> 1; s > 0; s >>= 1) {
      if (t < s) {
        float v2 = rv[t + s]; int i2 = ri[t + s];
        if (v2 > rv[t] || (v2 == rv[t] && i2 < ri[t])) { rv[t] = v2; ri[t] = i2; }
      }
      __syncthreads();
    }
    last = ri[0];
    if (t == last) {
      float* o = qsel + (size_t)(b * NP + j) * 3;
      o[0] = px; o[1] = py; o[2] = pz;
    }
    __syncthreads();
  }
}

// ---------------------------------------------------------------------------
// Ball query: first NS in-radius indices (ascending), pad with first (or 0).
// Also emits the sigmoid(|dp|) gate. One thread per query (b,i).
// ---------------------------------------------------------------------------
__global__ void k_ballquery(const float* __restrict__ pts, const float* __restrict__ qsel,
                            int* __restrict__ gidx, float* __restrict__ gate,
                            int B, int N, int NP, int NS, float radius) {
  const int qi = blockIdx.x * blockDim.x + threadIdx.x;
  if (qi >= B * NP) return;
  const int b = qi / NP;
  const float* q = qsel + (size_t)qi * 3;
  const float qx = q[0], qy = q[1], qz = q[2];
  const float r2 = radius * radius;
  const float* pb = pts + (size_t)b * N * 3;
  int g[32];
  int cnt = 0;
  for (int j = 0; j < N && cnt < NS; ++j) {
    float dx = pb[j * 3 + 0] - qx, dy = pb[j * 3 + 1] - qy, dz = pb[j * 3 + 2] - qz;
    if (dx * dx + dy * dy + dz * dz < r2) g[cnt++] = j;
  }
  const int pad = cnt ? g[0] : 0;
  for (int k = cnt; k < NS; ++k) g[k] = pad;
  const float inv_r = 1.0f / radius;
  for (int k = 0; k < NS; ++k) {
    const int j = g[k];
    float dx = (pb[j * 3 + 0] - qx) * inv_r;
    float dy = (pb[j * 3 + 1] - qy) * inv_r;
    float dz = (pb[j * 3 + 2] - qz) * inv_r;
    float d = sqrtf(dx * dx + dy * dy + dz * dz);
    gidx[(size_t)qi * NS + k] = j;
    gate[(size_t)qi * NS + k] = 1.0f / (1.0f + __expf(-d));
  }
}

// ---------------------------------------------------------------------------
// Assemble GEMM input X (Cpad x M, f16). Row c: c<3 -> dp, 3<=c<3+Cf -> feature,
// else zero (K padding). Coalesced along m.
// ---------------------------------------------------------------------------
__global__ void k_assembleX(const float* __restrict__ pts, const float* __restrict__ feat,
                            const float* __restrict__ qsel, const int* __restrict__ gidx,
                            _Float16* __restrict__ X,
                            int M, int N, int NP, int NS, int Cf, int featIsXyz, float inv_r) {
  const int m = blockIdx.x * blockDim.x + threadIdx.x;
  const int c = blockIdx.y;
  if (m >= M) return;
  float val = 0.0f;
  if (c < 3 + Cf) {
    const int qi = m / NS;
    const int b = qi / NP;
    const int j = gidx[m];
    if (c < 3) {
      val = (pts[(size_t)(b * N + j) * 3 + c] - qsel[(size_t)qi * 3 + c]) * inv_r;
    } else if (featIsXyz) {
      val = pts[(size_t)(b * N + j) * 3 + (c - 3)];
    } else {
      val = feat[(size_t)(b * Cf + (c - 3)) * N + j];
    }
  }
  X[(size_t)c * M + m] = (_Float16)val;
}

// ---------------------------------------------------------------------------
// Convert + zero-pad weights into A-fragment-swizzled f16 layout:
// Wf[((ot*nkc + kc)*32 + lane)*16 + e], e in fragment element order:
//   K = kc*32 + half*16 + e
// so the GEMM's A fragment is one contiguous 32-byte load per lane.
// ---------------------------------------------------------------------------
__global__ void k_padw_swz(const float* __restrict__ w, _Float16* __restrict__ wf,
                           int O, int C, int Cpad) {
  const int idx = blockIdx.x * blockDim.x + threadIdx.x;
  if (idx >= O * Cpad) return;
  const int e    = idx & 15;
  const int l    = (idx >> 4) & 31;
  const int rest = idx >> 9;
  const int nkc  = Cpad >> 5;
  const int kc   = rest % nkc;
  const int ot   = rest / nkc;
  const int half = l >> 4, mr = l & 15;
  const int kk   = (kc << 5) + half * 16 + e;
  const int row  = ot * 16 + mr;
  wf[idx] = (kk < C) ? (_Float16)w[row * C + kk] : (_Float16)0.0f;
}

// ---------------------------------------------------------------------------
// WMMA GEMM: Y(O x M, f32) = W(O x Cpad, f16, swizzled) * X(Cpad x M, f16).
// One block per 128-column m-panel. The FULL K panel (Cpad x 128, <=45KB) is
// staged transposed in LDS once, then the block loops over all O/16 output
// tiles -> X is read from HBM exactly once per GEMM.
// B fragment = one contiguous v16h LDS load (2x ds_load_b128).
// A fragment = one contiguous v16h global load from the swizzled weights.
// ---------------------------------------------------------------------------
__global__ void __launch_bounds__(256) k_gemm_wmma(const _Float16* __restrict__ Wf,
                                                   const _Float16* __restrict__ X,
                                                   float* __restrict__ Y,
                                                   int M, int Cpad, int O) {
  __shared__ _Float16 sXT[128 * 176];     // [col][k], stride = Cpad+16 (32B-aligned frags)
  const int tid  = threadIdx.x;
  const int wave = tid >> 5;
  const int lane = tid & 31;
  const int half = lane >> 4;
  const int mr   = lane & 15;
  const int mBase  = blockIdx.x * 128;
  const int stride = Cpad + 16;

  // Stage full K panel transposed: vectorized 16B global loads, coalesced.
  const int groups = (Cpad * 128) >> 3;   // 8-half groups
  for (int i = tid; i < groups; i += 256) {
    const int k  = i >> 4;                // source row (K)
    const int cg = (i & 15) << 3;         // starting column
    v8h v = *(const v8h*)(X + (size_t)k * M + mBase + cg);
    #pragma unroll
    for (int e = 0; e < 8; ++e)
      sXT[(cg + e) * stride + k] = v[e];
  }
  __syncthreads();

  const int nkc = Cpad >> 5;
  const int col = wave * 16 + mr;
  const _Float16* bbase = &sXT[col * stride + half * 16];

  for (int ot = 0; ot < (O >> 4); ++ot) {
    __builtin_prefetch(Wf + (((size_t)(ot + 1) * nkc) << 9), 0, 3);  // next tile's weights
    v8f acc = {};
    for (int kc = 0; kc < nkc; ++kc) {
      v16h a  = *(const v16h*)(Wf + ((size_t)(((ot * nkc + kc) << 5) + lane) << 4));
      v16h bf = *(const v16h*)(bbase + (kc << 5));
      acc = __builtin_amdgcn_wmma_f32_16x16x32_f16(false, a, false, bf,
                                                   (short)0, acc, false, false);
    }
    // C/D layout: VGPR r -> row r (lanes 0-15) / r+8 (lanes 16-31); N = lane&15
    #pragma unroll
    for (int r = 0; r < 8; ++r)
      Y[(size_t)(ot * 16 + r + 8 * half) * M + mBase + col] = acc[r];
  }
}

// ---------------------------------------------------------------------------
// Per-channel sum / sumsq for batchnorm (cross-batch stats).
// ---------------------------------------------------------------------------
__global__ void k_initf(float* __restrict__ p, int n) {
  const int i = blockIdx.x * blockDim.x + threadIdx.x;
  if (i < n) p[i] = 0.0f;
}

__global__ void k_stats(const float* __restrict__ Y, float* __restrict__ stats, int M, int O) {
  __shared__ float s1[256], s2[256];
  const int ch = blockIdx.y;
  const float* y = Y + (size_t)ch * M + (size_t)blockIdx.x * 4096;
  float a = 0.0f, b = 0.0f;
  for (int i = threadIdx.x; i < 4096; i += 256) { const float v = y[i]; a += v; b += v * v; }
  s1[threadIdx.x] = a; s2[threadIdx.x] = b;
  __syncthreads();
  for (int s = 128; s > 0; s >>= 1) {
    if (threadIdx.x < s) { s1[threadIdx.x] += s1[threadIdx.x + s]; s2[threadIdx.x] += s2[threadIdx.x + s]; }
    __syncthreads();
  }
  if (threadIdx.x == 0) { atomicAdd(&stats[ch], s1[0]); atomicAdd(&stats[O + ch], s2[0]); }
}

__global__ void k_finalize(const float* __restrict__ stats, const float* __restrict__ g,
                           const float* __restrict__ bias, float* __restrict__ ss,
                           int O, float invM) {
  const int c = blockIdx.x * blockDim.x + threadIdx.x;
  if (c >= O) return;
  const float mu  = stats[c] * invM;
  const float var = stats[O + c] * invM - mu * mu;
  const float sc  = g[c] * rsqrtf(var + 1e-5f);
  ss[c]     = sc;
  ss[O + c] = bias[c] - mu * sc;
}

// BN + leakyReLU + f16 convert -> next GEMM's X.
__global__ void k_act(const float* __restrict__ Y, const float* __restrict__ ss,
                      _Float16* __restrict__ X, int M, int O) {
  const int i = blockIdx.x * blockDim.x + threadIdx.x;
  if (i >= O * M) return;
  const int c = i / M;
  float v = Y[i] * ss[c] + ss[O + c];
  v = v > 0.0f ? v : 0.2f * v;
  X[i] = (_Float16)v;
}

// BN + leakyReLU + sigmoid-gate + max over nsample -> (B,O,NP) f32.
__global__ void k_pool(const float* __restrict__ Y, const float* __restrict__ gate,
                       const float* __restrict__ ss, float* __restrict__ out,
                       int M, int O, int NP, int NS, int B) {
  const int idx = blockIdx.x * blockDim.x + threadIdx.x;
  if (idx >= B * O * NP) return;
  const int i = idx % NP;
  const int rest = idx / NP;
  const int o = rest % O;
  const int b = rest / O;
  const float sc = ss[o], sh = ss[O + o];
  const int qbase = (b * NP + i) * NS;
  float mx = -1e30f;
  for (int k = 0; k < NS; ++k) {
    float v = Y[(size_t)o * M + qbase + k] * sc + sh;
    v = v > 0.0f ? v : 0.2f * v;
    mx = fmaxf(mx, gate[qbase + k] * v);
  }
  out[(size_t)(b * O + o) * NP + i] = mx;
}

// ---------------------------------------------------------------------------
extern "C" void kernel_launch(void* const* d_in, const int* in_sizes, int n_in,
                              void* d_out, int out_size, void* d_ws, size_t ws_size,
                              hipStream_t stream) {
  const float* xyz = (const float*)d_in[0];
  const float* w1a = (const float*)d_in[1];
  const float* g1a = (const float*)d_in[2];
  const float* b1a = (const float*)d_in[3];
  const float* w1b = (const float*)d_in[4];
  const float* g1b = (const float*)d_in[5];
  const float* b1b = (const float*)d_in[6];
  const float* w2a = (const float*)d_in[7];
  const float* g2a = (const float*)d_in[8];
  const float* b2a = (const float*)d_in[9];
  const float* w2b = (const float*)d_in[10];
  const float* g2b = (const float*)d_in[11];
  const float* b2b = (const float*)d_in[12];

  constexpr int B = 32, N1 = 512, NP1 = 256, NS1 = 16, N2 = 256, NP2 = 128, NS2 = 32;
  constexpr int M = B * NP1 * NS1;   // 131072 == B*NP2*NS2
  const float invM = 1.0f / (float)M;

  char* ws = (char*)d_ws;
  size_t off = 0;
  auto alloc = [&](size_t bytes) -> void* {
    void* p = ws + off;
    off = (off + bytes + 255) & ~(size_t)255;
    return p;
  };

  _Float16* wp1a = (_Float16*)alloc((size_t)64 * 32 * 2);
  _Float16* wp1b = (_Float16*)alloc((size_t)128 * 64 * 2);
  _Float16* wp2a = (_Float16*)alloc((size_t)128 * 160 * 2);
  _Float16* wp2b = (_Float16*)alloc((size_t)256 * 128 * 2);
  float* qsel1 = (float*)alloc((size_t)B * NP1 * 3 * 4);
  float* qsel2 = (float*)alloc((size_t)B * NP2 * 3 * 4);
  int*   gidx1 = (int*)  alloc((size_t)B * NP1 * NS1 * 4);
  int*   gidx2 = (int*)  alloc((size_t)B * NP2 * NS2 * 4);
  float* gate1 = (float*)alloc((size_t)B * NP1 * NS1 * 4);
  float* gate2 = (float*)alloc((size_t)B * NP2 * NS2 * 4);
  float* stats = (float*)alloc((size_t)2 * 256 * 4);
  float* ss    = (float*)alloc((size_t)2 * 256 * 4);
  float* f1    = (float*)alloc((size_t)B * 128 * NP1 * 4);
  _Float16* Xb = (_Float16*)alloc((size_t)160 * M * 2);   // reused by all 4 GEMMs
  float*    Yb = (float*)   alloc((size_t)256 * M * 4);   // reused by all 4 GEMMs

  // Weight prep into A-fragment-swizzled layout (tiny)
  k_padw_swz<<<(64 * 32 + 255) / 256, 256, 0, stream>>>(w1a, wp1a, 64, 6, 32);
  k_padw_swz<<<(128 * 64 + 255) / 256, 256, 0, stream>>>(w1b, wp1b, 128, 64, 64);
  k_padw_swz<<<(128 * 160 + 255) / 256, 256, 0, stream>>>(w2a, wp2a, 128, 131, 160);
  k_padw_swz<<<(256 * 128 + 255) / 256, 256, 0, stream>>>(w2b, wp2b, 256, 128, 128);

  // ----- Stage 1 -----
  k_shift_fps<N1, NP1><<<B, N1, 0, stream>>>(xyz, qsel1);
  k_ballquery<<<(B * NP1 + 255) / 256, 256, 0, stream>>>(xyz, qsel1, gidx1, gate1,
                                                         B, N1, NP1, NS1, 0.15f);
  k_assembleX<<<dim3(M / 256, 32), 256, 0, stream>>>(xyz, xyz, qsel1, gidx1, Xb,
                                                     M, N1, NP1, NS1, 3, 1, 1.0f / 0.15f);
  // conv 6->64
  k_gemm_wmma<<<M / 128, 256, 0, stream>>>(wp1a, Xb, Yb, M, 32, 64);
  k_initf<<<2, 256, 0, stream>>>(stats, 512);
  k_stats<<<dim3(M / 4096, 64), 256, 0, stream>>>(Yb, stats, M, 64);
  k_finalize<<<1, 64, 0, stream>>>(stats, g1a, b1a, ss, 64, invM);
  k_act<<<(64 * M + 255) / 256, 256, 0, stream>>>(Yb, ss, Xb, M, 64);
  // conv 64->128
  k_gemm_wmma<<<M / 128, 256, 0, stream>>>(wp1b, Xb, Yb, M, 64, 128);
  k_initf<<<2, 256, 0, stream>>>(stats, 512);
  k_stats<<<dim3(M / 4096, 128), 256, 0, stream>>>(Yb, stats, M, 128);
  k_finalize<<<1, 128, 0, stream>>>(stats, g1b, b1b, ss, 128, invM);
  k_pool<<<(B * 128 * NP1 + 255) / 256, 256, 0, stream>>>(Yb, gate1, ss, f1,
                                                          M, 128, NP1, NS1, B);

  // ----- Stage 2 -----
  k_shift_fps<N2, NP2><<<B, N2, 0, stream>>>(qsel1, qsel2);
  k_ballquery<<<(B * NP2 + 255) / 256, 256, 0, stream>>>(qsel1, qsel2, gidx2, gate2,
                                                         B, N2, NP2, NS2, 0.3f);
  k_assembleX<<<dim3(M / 256, 160), 256, 0, stream>>>(qsel1, f1, qsel2, gidx2, Xb,
                                                      M, N2, NP2, NS2, 128, 0, 1.0f / 0.3f);
  // conv 131->128
  k_gemm_wmma<<<M / 128, 256, 0, stream>>>(wp2a, Xb, Yb, M, 160, 128);
  k_initf<<<2, 256, 0, stream>>>(stats, 512);
  k_stats<<<dim3(M / 4096, 128), 256, 0, stream>>>(Yb, stats, M, 128);
  k_finalize<<<1, 128, 0, stream>>>(stats, g2a, b2a, ss, 128, invM);
  k_act<<<(128 * M + 255) / 256, 256, 0, stream>>>(Yb, ss, Xb, M, 128);
  // conv 128->256
  k_gemm_wmma<<<M / 128, 256, 0, stream>>>(wp2b, Xb, Yb, M, 128, 256);
  k_initf<<<2, 256, 0, stream>>>(stats, 512);
  k_stats<<<dim3(M / 4096, 256), 256, 0, stream>>>(Yb, stats, M, 256);
  k_finalize<<<1, 256, 0, stream>>>(stats, g2b, b2b, ss, 256, invM);
  k_pool<<<(B * 256 * NP2 + 255) / 256, 256, 0, stream>>>(Yb, gate2, ss, (float*)d_out,
                                                          M, 256, NP2, NS2, B);
}
